// GCN_7524782702918
// MI455X (gfx1250) — matrix operand order repaired and checked by reading, not compiled, not run on hardware
//
#include <hip/hip_runtime.h>
#include <hip/hip_bf16.h>

// GCN 3-layer inference for MI455X (gfx1250), wave32, WMMA f16->f32.
//
//   g = dinv .* (x @ W)
//   out[d] = dinv[d] * ( sum_{e: s->d} w_e * g[s] + g[d] ) + b
//   dinv = rsqrt(1 + sum_{e: ->d} w_e)   (self-loop weight 1), shared by all layers.
//
// X is pre-converted/padded to f16 once; W is pre-packed into per-lane WMMA
// B-fragments, so the GEMM inner loop is guard-free: 4x 16B vector loads + 1 WMMA.

typedef _Float16 v16h __attribute__((ext_vector_type(16)));
typedef _Float16 v8h  __attribute__((ext_vector_type(8)));
typedef float    v8f  __attribute__((ext_vector_type(8)));

#define F_IN1 269
#define KPAD1 288   // 269 padded up to multiple of 32

// ---------------- degree / dinv ----------------
__global__ void k_init_deg(float* deg, int n) {
    int i = blockIdx.x * blockDim.x + threadIdx.x;
    if (i < n) deg[i] = 1.0f;                 // self-loop weight
}

__global__ void k_accum_deg(const int* __restrict__ dst, const float* __restrict__ ew,
                            float* deg, int E) {
    int e = blockIdx.x * blockDim.x + threadIdx.x;
    if (e < E)
        __hip_atomic_fetch_add(&deg[dst[e]], ew[e], __ATOMIC_RELAXED, __HIP_MEMORY_SCOPE_AGENT);
}

__global__ void k_finish_deg(float* deg, int n) {
    int i = blockIdx.x * blockDim.x + threadIdx.x;
    if (i < n) {
        float d = deg[i];
        deg[i] = (d > 0.0f) ? rsqrtf(d) : 0.0f;
    }
}

// ---------------- X: f32 [N,269] -> f16 [N,288] zero-padded ----------------
__global__ void k_cvt_pad_x(const float* __restrict__ x, _Float16* __restrict__ xh, int K) {
    int row = blockIdx.x;
    int col = threadIdx.x;                    // blockDim.x == KPAD1
    xh[(size_t)row * KPAD1 + col] = (col < K) ? (_Float16)x[(size_t)row * K + col] : (_Float16)0.0f;
}

// ---------------- pack W [K,N] f32 -> per-lane WMMA B fragments (f16) -------
// Bp[((kt*ntiles + nt)*32 + lane)*16 + j] = W[kt*32 + (lane>>4)*16 + j][nt*16 + (lane&15)]
__global__ void k_pack_b(const float* __restrict__ W, _Float16* __restrict__ Bp,
                         int K, int N, int Kpad) {
    int idx = blockIdx.x * blockDim.x + threadIdx.x;
    int total = (Kpad >> 5) * (N >> 4) * 32 * 16;
    if (idx >= total) return;
    int j    = idx & 15;
    int lane = (idx >> 4) & 31;
    int t    = idx >> 9;                      // kt*ntiles + nt
    int ntiles = N >> 4;
    int kt = t / ntiles, nt = t - kt * ntiles;
    int kb  = (kt << 5) + ((lane >> 4) << 4) + j;
    int col = (nt << 4) + (lane & 15);
    float v = (kb < K) ? W[(size_t)kb * N + col] : 0.0f;
    Bp[(size_t)idx] = (_Float16)v;
}

// ---------------- WMMA GEMM: G[M,N] = dinv .* (Xh[M,Kpad] @ W)  ----------------
// One wave per 16x16 output tile; guard-free inner loop. M%16==0, N%16==0, Kpad%32==0.
__launch_bounds__(128)
__global__ void k_gemm_wmma(const _Float16* __restrict__ Xh, const _Float16* __restrict__ Bp,
                            const float* __restrict__ dinv, float* __restrict__ G,
                            int M, int Kpad, int N) {
    int wid  = (int)((blockIdx.x * 128u + threadIdx.x) >> 5);
    int lane = threadIdx.x & 31;
    int ntiles = N >> 4;
    int tiles  = (M >> 4) * ntiles;
    if (wid >= tiles) return;                 // uniform per wave: EXEC stays all-ones
    int mt = wid / ntiles;
    int nt = wid - mt * ntiles;
    int half = lane >> 4;
    int l16  = lane & 15;
    int row  = (mt << 4) + l16;
    int col  = (nt << 4) + l16;

    // A: one contiguous 16-half K run per lane: [k0+half*16 .. +15]
    const _Float16* xr  = Xh + (size_t)row * Kpad + half * 16;
    // B: this lane's packed 16-half fragment, advancing one k-tile per step
    const _Float16* bpl = Bp + ((size_t)nt * 32 + lane) * 16;
    size_t bstride = (size_t)ntiles * 32 * 16;

    v8f acc = {0.f, 0.f, 0.f, 0.f, 0.f, 0.f, 0.f, 0.f};
    int ktiles = Kpad >> 5;
#pragma unroll 2
    for (int kt = 0; kt < ktiles; ++kt) {
        v8h alo = *(const v8h*)(xr + (size_t)kt * 32);
        v8h ahi = *(const v8h*)(xr + (size_t)kt * 32 + 8);
        v16h a = __builtin_shufflevector(alo, ahi,
                                         0, 1, 2, 3, 4, 5, 6, 7,
                                         8, 9, 10, 11, 12, 13, 14, 15);
        v16h b = *(const v16h*)(bpl + (size_t)kt * bstride);
        acc = __builtin_amdgcn_wmma_f32_16x16x32_f16(
            /*neg_a=*/false, a, /*neg_b=*/false, b,
            /*c_mod=*/(short)0, acc, /*reuse_a=*/false, /*reuse_b=*/false);
    }
    // C/D layout: VGPR r holds rows r (lanes 0-15) and r+8 (lanes 16-31); fuse dinv scale.
#pragma unroll
    for (int r = 0; r < 8; ++r) {
        int m = (mt << 4) + r + half * 8;
        G[(size_t)m * N + col] = dinv[m] * acc[r];
    }
}

// ---------------- edge scatter: acc[d,:] += w * g[s,:] (float4 + f32 atomics) -----
__global__ void k_scatter4(const int* __restrict__ src, const int* __restrict__ dst,
                           const float* __restrict__ ew, const float* __restrict__ g,
                           float* __restrict__ acc, int total, int logFq) {
    int idx = blockIdx.x * blockDim.x + threadIdx.x;
    if (idx >= total) return;
    int e = idx >> logFq;
    int q = idx & ((1 << logFq) - 1);
    int s = src[e], d = dst[e];
    float w = ew[e];
    int F = 4 << logFq;
    const float4 gv = ((const float4*)(g + (size_t)s * F))[q];
    float* ap = acc + (size_t)d * F + q * 4;
    __hip_atomic_fetch_add(ap + 0, w * gv.x, __ATOMIC_RELAXED, __HIP_MEMORY_SCOPE_AGENT);
    __hip_atomic_fetch_add(ap + 1, w * gv.y, __ATOMIC_RELAXED, __HIP_MEMORY_SCOPE_AGENT);
    __hip_atomic_fetch_add(ap + 2, w * gv.z, __ATOMIC_RELAXED, __HIP_MEMORY_SCOPE_AGENT);
    __hip_atomic_fetch_add(ap + 3, w * gv.w, __ATOMIC_RELAXED, __HIP_MEMORY_SCOPE_AGENT);
}

__global__ void k_scatter1(const int* __restrict__ src, const int* __restrict__ dst,
                           const float* __restrict__ ew, const float* __restrict__ g,
                           float* __restrict__ acc, int E) {
    int e = blockIdx.x * blockDim.x + threadIdx.x;
    if (e < E)
        __hip_atomic_fetch_add(&acc[dst[e]], ew[e] * g[src[e]],
                               __ATOMIC_RELAXED, __HIP_MEMORY_SCOPE_AGENT);
}

// ---------------- xnext = f16( relu( dinv .* (acc + g) + b ) ) ----------------
__global__ void k_finalize_h(const float* __restrict__ acc, const float* __restrict__ g,
                             const float* __restrict__ dinv, const float* __restrict__ bias,
                             _Float16* __restrict__ outh, int total, int logF) {
    int idx = blockIdx.x * blockDim.x + threadIdx.x;
    if (idx < total) {
        int i = idx >> logF;
        int f = idx & ((1 << logF) - 1);
        float v = dinv[i] * (acc[idx] + g[idx]) + bias[f];
        outh[idx] = (_Float16)fmaxf(v, 0.0f);
    }
}

// ---------------- out = dinv .* (acc + g) + b  (f32, no relu; F==1) ----------------
__global__ void k_finalize_out(const float* __restrict__ acc, const float* __restrict__ g,
                               const float* __restrict__ dinv, const float* __restrict__ bias,
                               float* __restrict__ out, int n) {
    int i = blockIdx.x * blockDim.x + threadIdx.x;
    if (i < n) out[i] = dinv[i] * (acc[i] + g[i]) + bias[0];
}

// ---------------- layer 3 GEMV (Fout = 1): g[i] = dinv[i] * (x3[i,:] . W3) --------
__global__ void k_gemv32h(const _Float16* __restrict__ X, const float* __restrict__ W,
                          const float* __restrict__ dinv, float* __restrict__ g, int n) {
    int i = blockIdx.x * blockDim.x + threadIdx.x;
    if (i < n) {
        const _Float16* xr = X + (size_t)i * 32;
        float s = 0.0f;
#pragma unroll
        for (int k = 0; k < 32; ++k) s += (float)xr[k] * W[k];
        g[i] = dinv[i] * s;
    }
}

static inline dim3 gdim(long total, int block) {
    return dim3((unsigned)((total + block - 1) / block));
}
static inline size_t al256(size_t x) { return (x + 255) & ~(size_t)255; }

extern "C" void kernel_launch(void* const* d_in, const int* in_sizes, int n_in,
                              void* d_out, int out_size, void* d_ws, size_t ws_size,
                              hipStream_t stream) {
    const float* x  = (const float*)d_in[0];
    const int*   ei = (const int*)  d_in[1];
    const float* ew = (const float*)d_in[2];
    const float* W1 = (const float*)d_in[3];
    const float* b1 = (const float*)d_in[4];
    const float* W2 = (const float*)d_in[5];
    const float* b2 = (const float*)d_in[6];
    const float* W3 = (const float*)d_in[7];
    const float* b3 = (const float*)d_in[8];
    float* out = (float*)d_out;

    const int N = in_sizes[0] / F_IN1;   // 100000 (divisible by 16)
    const int E = in_sizes[2];           // 1600000
    const int* src = ei;
    const int* dst = ei + E;

    // ---- workspace layout (~110 MB), all regions rewritten before read each call ----
    char* p = (char*)d_ws;
    size_t off = 0;
    float* dinv = (float*)(p + off);              off = al256(off + (size_t)N * 4);
    const size_t bp1_elems = (KPAD1 / 32) * (64 / 16) * 32 * 16;   // 18432
    const size_t bp2_elems = (64 / 32) * (32 / 16) * 32 * 16;      //  2048
    _Float16* Bp1 = (_Float16*)(p + off);         off = al256(off + bp1_elems * 2);
    _Float16* Bp2 = (_Float16*)(p + off);         off = al256(off + bp2_elems * 2);
    _Float16* Xh  = (_Float16*)(p + off);         off = al256(off + (size_t)N * KPAD1 * 2);
    float* g   = (float*)(p + off);               off = al256(off + (size_t)N * 64 * 4);
    float* acc = (float*)(p + off);               off = al256(off + (size_t)N * 64 * 4);
    // f16 activations reuse the Xh region (dead after its GEMM consumes it)
    _Float16* x2h = Xh;     // N x 64 f16
    _Float16* x3h = Xh;     // N x 32 f16

    const dim3 B256(256), B128(128);

    // dinv (shared by all 3 layers)
    k_init_deg  <<<gdim(N, 256), B256, 0, stream>>>(dinv, N);
    k_accum_deg <<<gdim(E, 256), B256, 0, stream>>>(dst, ew, dinv, E);
    k_finish_deg<<<gdim(N, 256), B256, 0, stream>>>(dinv, N);

    // pre-convert X, pre-pack weights
    k_cvt_pad_x<<<dim3((unsigned)N), dim3(KPAD1), 0, stream>>>(x, Xh, F_IN1);
    k_pack_b<<<gdim((long)bp1_elems, 256), B256, 0, stream>>>(W1, Bp1, F_IN1, 64, KPAD1);
    k_pack_b<<<gdim((long)bp2_elems, 256), B256, 0, stream>>>(W2, Bp2, 64, 32, 64);

    // ---- layer 1: 269 -> 64, relu ----
    {
        long tiles = (long)(N / 16) * (64 / 16);
        hipMemsetAsync(acc, 0, (size_t)N * 64 * 4, stream);
        k_gemm_wmma<<<gdim(tiles * 32, 128), B128, 0, stream>>>(Xh, Bp1, dinv, g, N, KPAD1, 64);
        k_scatter4<<<gdim((long)E * 16, 256), B256, 0, stream>>>(src, dst, ew, g, acc, E * 16, 4);
        k_finalize_h<<<gdim((long)N * 64, 256), B256, 0, stream>>>(acc, g, dinv, b1, x2h, N * 64, 6);
    }
    // ---- layer 2: 64 -> 32, relu ----
    {
        long tiles = (long)(N / 16) * (32 / 16);
        hipMemsetAsync(acc, 0, (size_t)N * 32 * 4, stream);
        k_gemm_wmma<<<gdim(tiles * 32, 128), B128, 0, stream>>>(x2h, Bp2, dinv, g, N, 64, 32);
        k_scatter4<<<gdim((long)E * 8, 256), B256, 0, stream>>>(src, dst, ew, g, acc, E * 8, 3);
        k_finalize_h<<<gdim((long)N * 32, 256), B256, 0, stream>>>(acc, g, dinv, b2, x3h, N * 32, 5);
    }
    // ---- layer 3: 32 -> 1, no relu, write d_out ----
    {
        hipMemsetAsync(acc, 0, (size_t)N * 4, stream);
        k_gemv32h<<<gdim(N, 256), B256, 0, stream>>>(x3h, W3, dinv, g, N);
        k_scatter1<<<gdim(E, 256), B256, 0, stream>>>(src, dst, ew, g, acc, E);
        k_finalize_out<<<gdim(N, 256), B256, 0, stream>>>(acc, g, dinv, b3, out, N);
    }
}
